// PyramidAxialEncoder_84963043050027
// MI455X (gfx1250) — matrix operand order repaired and checked by reading, not compiled, not run on hardware
//
#include <hip/hip_runtime.h>
#include <hip/hip_bf16.h>
#include <math.h>

// ---------------------------------------------------------------------------
// CDNA5 (gfx1250) implementation of the PyramidAxialEncoder reference.
// All dense linear algebra goes through v_wmma_f32_16x16x32_f16 (wave32).
// GEMM v2: 64x64 block tile, each wave = 16x32 (1 A-frag reused by 2 WMMAs),
// plus global_prefetch of the next A K-chunk.
// ---------------------------------------------------------------------------

typedef _Float16 v8h  __attribute__((ext_vector_type(8)));
typedef _Float16 v16h __attribute__((ext_vector_type(16)));
typedef float    v8f  __attribute__((ext_vector_type(8)));

#define N_CAM   6
#define DIMC    128
#define FH      32
#define FW      64
#define HB      128
#define WB      128
#define NS      (HB*WB)        // 16384 BEV positions
#define NPIX    (FH*FW)        // 2048 image-plane positions
#define NKV     (N_CAM*NPIX)   // 12288 key/value tokens
#define NQ1     (N_CAM*NS)     // 98304 attn1 query tokens
#define SCALE_ATT 0.17677669529663687f   // 32^-0.5
#define EPSN    1e-7f

// ---------------- WMMA helpers -------------------------------------------

__device__ __forceinline__ v8f wmma_f16(v16h a, v16h b, v8f c) {
  return __builtin_amdgcn_wmma_f32_16x16x32_f16(false, a, false, b, (short)0, c,
                                                false, false);
}

// A fragment from a row-major [16][K>=32] LDS tile (16-bit A layout:
// lanes 0-15 rows M, K 0-7 / 16-23; lanes 16-31 rows M, K 8-15 / 24-31).
__device__ __forceinline__ v16h lds_fragA(const _Float16* base, int stride) {
  int lane = threadIdx.x & 31;
  int m = lane & 15, hi = lane >> 4;
  const _Float16* rp = base + m * stride;
  v8h lo = *(const v8h*)(rp + hi * 8);
  v8h hh = *(const v8h*)(rp + 16 + hi * 8);
  return __builtin_shufflevector(lo, hh, 0, 1, 2, 3, 4, 5, 6, 7, 8, 9, 10, 11,
                                 12, 13, 14, 15);
}

// B fragment from a *transposed* LDS tile stored [n][k] (row n = output col):
// lanes 0-15 col n with K 0-7 / 16-23, lanes 16-31 col n with K 8-15 / 24-31.
__device__ __forceinline__ v16h lds_fragB(const _Float16* baseT, int stride) {
  int lane = threadIdx.x & 31;
  int n = lane & 15, hi = lane >> 4;
  const _Float16* rp = baseT + n * stride + hi * 8;
  v8h lo = *(const v8h*)(rp);
  v8h hh = *(const v8h*)(rp + 16);
  return __builtin_shufflevector(lo, hh, 0, 1, 2, 3, 4, 5, 6, 7, 8, 9, 10, 11,
                                 12, 13, 14, 15);
}

__device__ __forceinline__ float gelu_exact(float x) {
  return 0.5f * x * (1.0f + erff(x * 0.70710678118654752f));
}

// window-unpartition row remap: token index t = l*256 + qq  ->  spatial h*W+w
__device__ __forceinline__ int remap_row(int t, int mode) {
  if (mode == 0) return t;
  int l = t >> 8, qq = t & 255;
  int X = l >> 3, Y = l & 7, i = qq >> 4, j = qq & 15;
  int h, w;
  if (mode == 1) { h = X * 16 + i; w = Y * 16 + j; }   // attn1 (x w1)(y w2)
  else           { h = i * 8 + X;  w = j * 8 + Y;  }   // attn2 (w1 x)(w2 y)
  return h * WB + w;
}

__device__ __forceinline__ float blk128_sum(float v, float* sh) {
#pragma unroll
  for (int m = 16; m >= 1; m >>= 1) v += __shfl_xor(v, m, 32);
  if ((threadIdx.x & 31) == 0) sh[threadIdx.x >> 5] = v;
  __syncthreads();
  float r = sh[0] + sh[1] + sh[2] + sh[3];
  __syncthreads();
  return r;
}

// ---------------- small elementwise kernels --------------------------------

// f32 -> f16 (optionally transposed) copy; used for weights and abar.
__global__ __launch_bounds__(256) void cvt_f16_kernel(
    const float* __restrict__ src, _Float16* __restrict__ dst, int R, int C,
    int trans) {
  int idx = blockIdx.x * 256 + threadIdx.x;
  if (idx >= R * C) return;
  int r = idx / C, c = idx - r * C;
  float v = trans ? src[(size_t)c * R + r] : src[idx];
  dst[idx] = (_Float16)v;
}

// c_embed[n][d] = cam_w[d] . E_inv[n][:, 3]
__global__ __launch_bounds__(256) void cam_embed_kernel(
    const float* __restrict__ cam_w, const float* __restrict__ E_inv,
    float* __restrict__ c_emb) {
  int idx = blockIdx.x * 256 + threadIdx.x;
  if (idx >= N_CAM * DIMC) return;
  int n = idx >> 7, d = idx & 127;
  float s = 0.f;
#pragma unroll
  for (int c = 0; c < 4; c++) s += cam_w[d * 4 + c] * E_inv[n * 16 + c * 4 + 3];
  c_emb[idx] = s;
}

// img_embed[(n*2048+p)][d]  (normalized over d)
__global__ __launch_bounds__(128) void img_embed_kernel(
    const float* __restrict__ I_inv, const float* __restrict__ E_inv,
    const float* __restrict__ img_w, const float* __restrict__ c_emb,
    float* __restrict__ img_embed) {
  __shared__ float sh[4];
  int row = blockIdx.x;
  int n = row >> 11, p = row & 2047;
  int d = threadIdx.x;
  int py = p >> 6, px = p & 63;
  float u = px * (1024.0f / 63.0f);
  float v = py * (512.0f / 31.0f);
  const float* Ii = I_inv + n * 9;
  float c0 = Ii[0] * u + Ii[1] * v + Ii[2];
  float c1 = Ii[3] * u + Ii[4] * v + Ii[5];
  float c2 = Ii[6] * u + Ii[7] * v + Ii[8];
  const float* Ei = E_inv + n * 16;
  float r0 = Ei[0] * c0 + Ei[1] * c1 + Ei[2] * c2 + Ei[3];
  float r1 = Ei[4] * c0 + Ei[5] * c1 + Ei[6] * c2 + Ei[7];
  float r2 = Ei[8] * c0 + Ei[9] * c1 + Ei[10] * c2 + Ei[11];
  float r3 = Ei[12] * c0 + Ei[13] * c1 + Ei[14] * c2 + Ei[15];
  float e = img_w[d * 4 + 0] * r0 + img_w[d * 4 + 1] * r1 +
            img_w[d * 4 + 2] * r2 + img_w[d * 4 + 3] * r3 -
            c_emb[n * 128 + d];
  float ss = blk128_sum(e * e, sh);
  e = e / (sqrtf(ss) + EPSN);
  img_embed[(size_t)row * 128 + d] = e;
}

// BN(eval) + ReLU + f16:  act16[(n*2048+p)*128 + c]
__global__ __launch_bounds__(256) void bn_relu_kernel(
    const float* __restrict__ feat, const float* __restrict__ g,
    const float* __restrict__ b, const float* __restrict__ m,
    const float* __restrict__ v, _Float16* __restrict__ act16) {
  int idx = blockIdx.x * 256 + threadIdx.x;
  if (idx >= NKV * DIMC) return;
  int pos = idx >> 7, c = idx & 127;
  int n = pos >> 11, p = pos & 2047;
  float f = feat[((size_t)(n * 128 + c)) * 2048 + p];
  float xn = (f - m[c]) * (g[c] * __frsqrt_rn(v[c] + 1e-5f)) + b[c];
  act16[idx] = (_Float16)fmaxf(xn, 0.0f);
}

// fused: bev_embed + x  -> LayerNorm -> f16 (row = n*16384 + s)
__global__ __launch_bounds__(128) void query_ln_kernel(
    const float* __restrict__ world, const float* __restrict__ bev_w,
    const float* __restrict__ bev_b, const float* __restrict__ c_emb,
    const float* __restrict__ x, const float* __restrict__ lng,
    const float* __restrict__ lnb, _Float16* __restrict__ out) {
  __shared__ float sh[4];
  int row = blockIdx.x;
  int n = row / NS, s = row - n * NS;
  int d = threadIdx.x;
  float wv0 = world[s], wv1 = world[NS + s];
  float e = bev_w[d * 2 + 0] * wv0 + bev_w[d * 2 + 1] * wv1 + bev_b[d] -
            c_emb[n * 128 + d];
  float ss = blk128_sum(e * e, sh);
  float q = e / (sqrtf(ss) + EPSN) + x[(size_t)d * NS + s];
  float mean = blk128_sum(q, sh) * (1.0f / 128.0f);
  float dv = q - mean;
  float var = blk128_sum(dv * dv, sh) * (1.0f / 128.0f);
  float y = dv * __frsqrt_rn(var + 1e-5f) * lng[d] + lnb[d];
  out[(size_t)row * 128 + d] = (_Float16)y;
}

// LayerNorm f32 rows -> f16 rows
__global__ __launch_bounds__(128) void ln128_kernel(
    const float* __restrict__ xin, const float* __restrict__ g,
    const float* __restrict__ b, _Float16* __restrict__ out) {
  __shared__ float sh[4];
  int row = blockIdx.x, d = threadIdx.x;
  float v = xin[(size_t)row * 128 + d];
  float mean = blk128_sum(v, sh) * (1.0f / 128.0f);
  float dv = v - mean;
  float var = blk128_sum(dv * dv, sh) * (1.0f / 128.0f);
  out[(size_t)row * 128 + d] =
      (_Float16)(dv * __frsqrt_rn(var + 1e-5f) * g[d] + b[d]);
}

// final LayerNorm, write transposed f32 output (dim, H, W)
__global__ __launch_bounds__(128) void final_ln_kernel(
    const float* __restrict__ xin, const float* __restrict__ g,
    const float* __restrict__ b, float* __restrict__ out) {
  __shared__ float sh[4];
  int s = blockIdx.x, d = threadIdx.x;
  float v = xin[(size_t)s * 128 + d];
  float mean = blk128_sum(v, sh) * (1.0f / 128.0f);
  float dv = v - mean;
  float var = blk128_sum(dv * dv, sh) * (1.0f / 128.0f);
  out[(size_t)d * NS + s] = dv * __frsqrt_rn(var + 1e-5f) * g[d] + b[d];
}

// ---------------- generic WMMA GEMM ---------------------------------------
// C[M][N] = A_f16[M][K] @ B_f16[K][N]  (+bias)(gelu)(+f32 addsrc)(row remap)
// Block tile 64x64 (8 waves, each 16x32 = one A-frag feeding 2 WMMAs),
// K stepped by 32, next-chunk A prefetched (global_prefetch_b8).
__global__ __launch_bounds__(256) void gemm_wmma_kernel(
    const _Float16* __restrict__ A, const _Float16* __restrict__ Bw,
    float* __restrict__ outF, _Float16* __restrict__ outH,
    const float* __restrict__ bias, const float* __restrict__ addsrc, int M,
    int N, int K, int remapMode, int actMode) {
  __shared__ _Float16 As[64 * 40];   // [m][k] row-major
  __shared__ _Float16 Bt[64 * 40];   // [n][k] transposed
  int tid = threadIdx.x;
  int wave = tid >> 5, lane = tid & 31;
  int wr = wave >> 1, wc = wave & 1;           // 4x2 waves, each 16x32
  int m0 = blockIdx.x * 64, n0 = blockIdx.y * 64;
  v8f acc0 = {0.f, 0.f, 0.f, 0.f, 0.f, 0.f, 0.f, 0.f};
  v8f acc1 = {0.f, 0.f, 0.f, 0.f, 0.f, 0.f, 0.f, 0.f};

  int ar = tid >> 2, ac8 = (tid & 3) * 8;      // A staging coords
  int bk = tid >> 3, bn8 = (tid & 7) * 8;      // B staging coords

  for (int k0 = 0; k0 < K; k0 += 32) {
    {  // stage A 64x32 (v8h, 16B aligned) + prefetch next K-chunk
      const _Float16* ap = &A[(size_t)(m0 + ar) * K + k0 + ac8];
      *(v8h*)&As[ar * 40 + ac8] = *(const v8h*)ap;
      if (k0 + 32 < K) __builtin_prefetch(ap + 32, 0, 1);
    }
    {  // stage B 32x64, transposed to [n][k]
      v8h bv = *(const v8h*)&Bw[(size_t)(k0 + bk) * N + n0 + bn8];
#pragma unroll
      for (int q = 0; q < 8; q++) Bt[(bn8 + q) * 40 + bk] = bv[q];
    }
    __syncthreads();
    v16h af  = lds_fragA(&As[(wr * 16) * 40], 40);
    v16h bf0 = lds_fragB(&Bt[(wc * 32) * 40], 40);
    v16h bf1 = lds_fragB(&Bt[(wc * 32 + 16) * 40], 40);
    acc0 = wmma_f16(af, bf0, acc0);
    acc1 = wmma_f16(af, bf1, acc1);
    __syncthreads();
  }

  int hi = lane >> 4, nn = lane & 15;
  int col0 = n0 + wc * 32 + nn;
  int col1 = col0 + 16;
  float bv0 = bias ? bias[col0] : 0.0f;
  float bv1 = bias ? bias[col1] : 0.0f;
#pragma unroll
  for (int r = 0; r < 8; r++) {
    int row = m0 + wr * 16 + r + 8 * hi;
    int orow = remap_row(row, remapMode);
    float v0 = acc0[r] + bv0;
    float v1 = acc1[r] + bv1;
    if (actMode == 1) { v0 = gelu_exact(v0); v1 = gelu_exact(v1); }
    if (addsrc) {
      v0 += addsrc[(size_t)orow * N + col0];
      v1 += addsrc[(size_t)orow * N + col1];
    }
    if (outH) {
      outH[(size_t)orow * N + col0] = (_Float16)v0;
      outH[(size_t)orow * N + col1] = (_Float16)v1;
    } else {
      outF[(size_t)orow * N + col0] = v0;
      outF[(size_t)orow * N + col1] = v1;
    }
  }
}

// ---------------- fused window attention core ------------------------------
// One block = (window l, head h, 128 query rows). K tokens = 192.
// S = q @ k^T (12 WMMAs), register softmax (width-16 shuffles),
// P@V via per-wave LDS P chunks (LDS in-order per wave, no barrier needed).
__global__ __launch_bounds__(256) void attn_core_kernel(
    const _Float16* __restrict__ qh, const _Float16* __restrict__ kh,
    const _Float16* __restrict__ vh, float* __restrict__ accOut,
    _Float16* __restrict__ aOut, int swapped, int nQB, float outScale) {
  __shared__ _Float16 Kt[192 * 40];   // [token][d]  (head slice)
  __shared__ _Float16 Vt[32 * 200];   // [d][token]  (transposed)
  __shared__ _Float16 Qs[128 * 40];   // [row][d]
  __shared__ _Float16 Pb[8 * 16 * 40];// per-wave P chunk

  int bid = blockIdx.x;
  int qb = bid % nQB;
  int t2 = bid / nQB;
  int head = t2 & 3;
  int l = t2 >> 2;
  int X = l >> 3, Y = l & 7;
  int tid = threadIdx.x, wave = tid >> 5, lane = tid & 31;
  int hi = lane >> 4, nn = lane & 15;

  // ---- stage K (row-major head slice) and V (transposed) ----
  if (tid < 192) {
    int n = tid >> 5, r5 = tid & 31, i = r5 >> 3, j = r5 & 7;
    int fh = swapped ? (i * 8 + X) : (X * 4 + i);
    int fw = swapped ? (j * 8 + Y) : (Y * 8 + j);
    size_t pos = (size_t)n * NPIX + fh * FW + fw;
    const v8h* kp = (const v8h*)&kh[pos * 128 + head * 32];
    const v8h* vp = (const v8h*)&vh[pos * 128 + head * 32];
    v8h k0 = kp[0], k1 = kp[1], k2 = kp[2], k3 = kp[3];
    *(v8h*)&Kt[tid * 40 + 0]  = k0;
    *(v8h*)&Kt[tid * 40 + 8]  = k1;
    *(v8h*)&Kt[tid * 40 + 16] = k2;
    *(v8h*)&Kt[tid * 40 + 24] = k3;
    v8h w0 = vp[0], w1 = vp[1], w2 = vp[2], w3 = vp[3];
#pragma unroll
    for (int d = 0; d < 8; d++) {
      Vt[(d + 0)  * 200 + tid] = w0[d];
      Vt[(d + 8)  * 200 + tid] = w1[d];
      Vt[(d + 16) * 200 + tid] = w2[d];
      Vt[(d + 24) * 200 + tid] = w3[d];
    }
  }
  // ---- stage Q (gather rows via window map) ----
  {
    int r = tid >> 1, part = tid & 1;
    int tq = qb * 128 + r;
    int grow;
    if (swapped) {
      int i = (tq & 255) >> 4, j = tq & 15;
      grow = (i * 8 + X) * WB + (j * 8 + Y);
    } else {
      int n = tq >> 8, qq = tq & 255, i = qq >> 4, j = qq & 15;
      grow = n * NS + (X * 16 + i) * WB + (Y * 16 + j);
    }
    const v8h* qp = (const v8h*)&qh[(size_t)grow * 128 + head * 32 + part * 16];
    *(v8h*)&Qs[r * 40 + part * 16 + 0] = qp[0];
    *(v8h*)&Qs[r * 40 + part * 16 + 8] = qp[1];
  }
  __syncthreads();

  // ---- S = Q K^T : 12 tiles of 16 key tokens ----
  v16h aq = lds_fragA(&Qs[(wave * 16) * 40], 40);
  v8f S[12];
#pragma unroll
  for (int t = 0; t < 12; t++) {
    v16h bk = lds_fragB(&Kt[(t * 16) * 40], 40);
    v8f z = {0.f, 0.f, 0.f, 0.f, 0.f, 0.f, 0.f, 0.f};
    S[t] = wmma_f16(aq, bk, z);
  }

  // ---- softmax over 192 keys (rows r+8*hi, cols across lanes 0..15) ----
#pragma unroll
  for (int r = 0; r < 8; r++) {
    float mv = -3.0e38f;
#pragma unroll
    for (int t = 0; t < 12; t++) mv = fmaxf(mv, S[t][r]);
#pragma unroll
    for (int m = 8; m >= 1; m >>= 1) mv = fmaxf(mv, __shfl_xor(mv, m, 32));
    float sum = 0.f;
#pragma unroll
    for (int t = 0; t < 12; t++) {
      float e = __expf(SCALE_ATT * (S[t][r] - mv));
      S[t][r] = e;
      sum += e;
    }
#pragma unroll
    for (int m = 8; m >= 1; m >>= 1) sum += __shfl_xor(sum, m, 32);
    float ri = 1.0f / sum;
#pragma unroll
    for (int t = 0; t < 12; t++) S[t][r] *= ri;
  }

  // ---- O = P @ V, K=192 in 6 chunks of 32 ----
  _Float16* pb = &Pb[wave * 640];
  v8f O0 = {0.f, 0.f, 0.f, 0.f, 0.f, 0.f, 0.f, 0.f};
  v8f O1 = {0.f, 0.f, 0.f, 0.f, 0.f, 0.f, 0.f, 0.f};
#pragma unroll
  for (int kc = 0; kc < 6; kc++) {
#pragma unroll
    for (int r = 0; r < 8; r++) {
      pb[(r + 8 * hi) * 40 + nn]      = (_Float16)S[2 * kc + 0][r];
      pb[(r + 8 * hi) * 40 + 16 + nn] = (_Float16)S[2 * kc + 1][r];
    }
    v16h ap = lds_fragA(pb, 40);
    v16h b0 = lds_fragB(&Vt[0 * 200 + kc * 32], 200);
    v16h b1 = lds_fragB(&Vt[16 * 200 + kc * 32], 200);
    O0 = wmma_f16(ap, b0, O0);
    O1 = wmma_f16(ap, b1, O1);
  }

  // ---- store ----
  int colb = head * 32;
#pragma unroll
  for (int r = 0; r < 8; r++) {
    int tq = qb * 128 + wave * 16 + r + 8 * hi;
    if (accOut) {  // attn1: view-mean via atomics into abar[l*256+qq]
      int row = l * 256 + (tq & 255);
      atomicAdd(&accOut[(size_t)row * 128 + colb + nn], O0[r] * outScale);
      atomicAdd(&accOut[(size_t)row * 128 + colb + 16 + nn], O1[r] * outScale);
    } else {       // attn2: direct f16 store, row = l*256 + tq
      int row = l * 256 + tq;
      aOut[(size_t)row * 128 + colb + nn] = (_Float16)O0[r];
      aOut[(size_t)row * 128 + colb + 16 + nn] = (_Float16)O1[r];
    }
  }
}

// ---------------------------------------------------------------------------

extern "C" void kernel_launch(void* const* d_in, const int* in_sizes, int n_in,
                              void* d_out, int out_size, void* d_ws,
                              size_t ws_size, hipStream_t stream) {
  (void)in_sizes; (void)n_in; (void)out_size; (void)ws_size;
  const float* x      = (const float*)d_in[0];
  const float* feat   = (const float*)d_in[1];
  const float* I_inv  = (const float*)d_in[2];
  const float* E_inv  = (const float*)d_in[3];
  const float* world  = (const float*)d_in[4];
  const float* cam_w  = (const float*)d_in[5];
  const float* img_w  = (const float*)d_in[6];
  const float* bev_w  = (const float*)d_in[7];
  const float* bev_b  = (const float*)d_in[8];
  const float* bn1_g  = (const float*)d_in[9];
  const float* bn1_b  = (const float*)d_in[10];
  const float* bn1_m  = (const float*)d_in[11];
  const float* bn1_v  = (const float*)d_in[12];
  const float* bn2_g  = (const float*)d_in[13];
  const float* bn2_b  = (const float*)d_in[14];
  const float* bn2_m  = (const float*)d_in[15];
  const float* bn2_v  = (const float*)d_in[16];
  const float* proj_w = (const float*)d_in[17];
  const float* lin_w  = (const float*)d_in[18];
  // attn1: 19..29, attn2: 30..40
  const float* a1[11]; const float* a2[11];
  for (int i = 0; i < 11; i++) { a1[i] = (const float*)d_in[19 + i];
                                 a2[i] = (const float*)d_in[30 + i]; }
  // [0]lnq_g [1]lnq_b [2]lnk_g [3]lnk_b [4]lnv_g [5]lnv_b
  // [6]Wq [7]Wk [8]Wv [9]Wo [10]bo
  const float* pre1_g = (const float*)d_in[41];
  const float* pre1_b = (const float*)d_in[42];
  const float* pre2_g = (const float*)d_in[43];
  const float* pre2_b = (const float*)d_in[44];
  const float* post_g = (const float*)d_in[45];
  const float* post_b = (const float*)d_in[46];
  const float* m1W1 = (const float*)d_in[47];
  const float* m1b1 = (const float*)d_in[48];
  const float* m1W2 = (const float*)d_in[49];
  const float* m1b2 = (const float*)d_in[50];
  const float* m2W1 = (const float*)d_in[51];
  const float* m2b1 = (const float*)d_in[52];
  const float* m2W2 = (const float*)d_in[53];
  const float* m2b2 = (const float*)d_in[54];

  // ------------- workspace layout (~150 MB) -------------
  size_t off = 0;
  auto alloc = [&](size_t bytes) -> void* {
    void* p = (void*)((char*)d_ws + off);
    off = (off + bytes + 255) & ~(size_t)255;
    return p;
  };
  float*    c_emb   = (float*)alloc((size_t)N_CAM * DIMC * 4);
  float*    imge    = (float*)alloc((size_t)NKV * DIMC * 4);
  float*    keyb    = (float*)alloc((size_t)NKV * DIMC * 4);
  float*    valb    = (float*)alloc((size_t)NKV * DIMC * 4);
  _Float16* act16   = (_Float16*)alloc((size_t)NKV * DIMC * 2);
  _Float16* wprojT  = (_Float16*)alloc(128 * 128 * 2);
  _Float16* wlinT   = (_Float16*)alloc(128 * 128 * 2);
  _Float16* wq1     = (_Float16*)alloc(128 * 128 * 2);
  _Float16* wk1     = (_Float16*)alloc(128 * 128 * 2);
  _Float16* wv1     = (_Float16*)alloc(128 * 128 * 2);
  _Float16* wo1     = (_Float16*)alloc(128 * 128 * 2);
  _Float16* wq2     = (_Float16*)alloc(128 * 128 * 2);
  _Float16* wk2     = (_Float16*)alloc(128 * 128 * 2);
  _Float16* wv2     = (_Float16*)alloc(128 * 128 * 2);
  _Float16* wo2     = (_Float16*)alloc(128 * 128 * 2);
  _Float16* w11     = (_Float16*)alloc(128 * 256 * 2);
  _Float16* w21     = (_Float16*)alloc(256 * 128 * 2);
  _Float16* w12     = (_Float16*)alloc(128 * 256 * 2);
  _Float16* w22     = (_Float16*)alloc(256 * 128 * 2);
  _Float16* lnq1    = (_Float16*)alloc((size_t)NQ1 * 128 * 2);
  _Float16* qh1     = (_Float16*)alloc((size_t)NQ1 * 128 * 2);
  _Float16* lnkv    = (_Float16*)alloc((size_t)NKV * 128 * 2);
  _Float16* khb     = (_Float16*)alloc((size_t)NKV * 128 * 2);
  _Float16* vhb     = (_Float16*)alloc((size_t)NKV * 128 * 2);
  float*    abar    = (float*)alloc((size_t)NS * 128 * 4);
  _Float16* abar16  = (_Float16*)alloc((size_t)NS * 128 * 2);
  float*    qhwd    = (float*)alloc((size_t)NS * 128 * 4);
  _Float16* lntmp   = (_Float16*)alloc((size_t)NS * 128 * 2);
  _Float16* hbuf    = (_Float16*)alloc((size_t)NS * 256 * 2);
  float*    xskip   = (float*)alloc((size_t)NS * 128 * 4);
  _Float16* qh2b    = (_Float16*)alloc((size_t)NS * 128 * 2);
  _Float16* a2b     = (_Float16*)alloc((size_t)NS * 128 * 2);
  float*    z2b     = (float*)alloc((size_t)NS * 128 * 4);
  float*    q2f     = (float*)alloc((size_t)NS * 128 * 4);

  hipMemsetAsync(abar, 0, (size_t)NS * 128 * 4, stream);

  // ------------- weight conversion to f16 -------------
  auto cvt = [&](const float* s, _Float16* dst, int R, int C, int tr) {
    cvt_f16_kernel<<<(R * C + 255) / 256, 256, 0, stream>>>(s, dst, R, C, tr);
  };
  cvt(proj_w, wprojT, 128, 128, 1);  // [in][out]
  cvt(lin_w,  wlinT,  128, 128, 1);
  cvt(a1[6], wq1, 128, 128, 0); cvt(a1[7], wk1, 128, 128, 0);
  cvt(a1[8], wv1, 128, 128, 0); cvt(a1[9], wo1, 128, 128, 0);
  cvt(a2[6], wq2, 128, 128, 0); cvt(a2[7], wk2, 128, 128, 0);
  cvt(a2[8], wv2, 128, 128, 0); cvt(a2[9], wo2, 128, 128, 0);
  cvt(m1W1, w11, 128, 256, 0); cvt(m1W2, w21, 256, 128, 0);
  cvt(m2W1, w12, 128, 256, 0); cvt(m2W2, w22, 256, 128, 0);

  auto gemm = [&](const _Float16* A, const _Float16* B, float* oF,
                  _Float16* oH, const float* bias, const float* add, int M,
                  int N, int K, int remap, int act) {
    gemm_wmma_kernel<<<dim3(M / 64, N / 64), 256, 0, stream>>>(
        A, B, oF, oH, bias, add, M, N, K, remap, act);
  };

  // ------------- geometry embeddings -------------
  cam_embed_kernel<<<3, 256, 0, stream>>>(cam_w, E_inv, c_emb);
  img_embed_kernel<<<NKV, 128, 0, stream>>>(I_inv, E_inv, img_w, c_emb, imge);

  // key = img_embed + proj(bn1relu(feat)); val = lin(bn2relu(feat))
  bn_relu_kernel<<<(NKV * 128) / 256, 256, 0, stream>>>(feat, bn1_g, bn1_b,
                                                        bn1_m, bn1_v, act16);
  gemm(act16, wprojT, keyb, nullptr, nullptr, imge, NKV, 128, 128, 0, 0);
  bn_relu_kernel<<<(NKV * 128) / 256, 256, 0, stream>>>(feat, bn2_g, bn2_b,
                                                        bn2_m, bn2_v, act16);
  gemm(act16, wlinT, valb, nullptr, nullptr, nullptr, NKV, 128, 128, 0, 0);

  // ------------- attention block 1 -------------
  query_ln_kernel<<<NQ1, 128, 0, stream>>>(world, bev_w, bev_b, c_emb, x,
                                           a1[0], a1[1], lnq1);
  gemm(lnq1, wq1, nullptr, qh1, nullptr, nullptr, NQ1, 128, 128, 0, 0);
  ln128_kernel<<<NKV, 128, 0, stream>>>(keyb, a1[2], a1[3], lnkv);
  gemm(lnkv, wk1, nullptr, khb, nullptr, nullptr, NKV, 128, 128, 0, 0);
  ln128_kernel<<<NKV, 128, 0, stream>>>(valb, a1[4], a1[5], lnkv);
  gemm(lnkv, wv1, nullptr, vhb, nullptr, nullptr, NKV, 128, 128, 0, 0);

  attn_core_kernel<<<64 * 4 * 12, 256, 0, stream>>>(
      qh1, khb, vhb, abar, nullptr, /*swapped=*/0, /*nQB=*/12, 1.0f / 6.0f);

  cvt_f16_kernel<<<(NS * 128 + 255) / 256, 256, 0, stream>>>(abar, abar16, NS,
                                                             128, 0);
  gemm(abar16, wo1, qhwd, nullptr, a1[10], nullptr, NS, 128, 128, /*remap*/1, 0);

  // MLP1 with residual
  ln128_kernel<<<NS, 128, 0, stream>>>(qhwd, pre1_g, pre1_b, lntmp);
  gemm(lntmp, w11, nullptr, hbuf, m1b1, nullptr, NS, 256, 128, 0, /*gelu*/1);
  gemm(hbuf, w21, xskip, nullptr, m1b2, qhwd, NS, 128, 256, 0, 0);

  // ------------- attention block 2 (query broadcast across views) ---------
  ln128_kernel<<<NS, 128, 0, stream>>>(xskip, a2[0], a2[1], lntmp);
  gemm(lntmp, wq2, nullptr, qh2b, nullptr, nullptr, NS, 128, 128, 0, 0);
  ln128_kernel<<<NKV, 128, 0, stream>>>(keyb, a2[2], a2[3], lnkv);
  gemm(lnkv, wk2, nullptr, khb, nullptr, nullptr, NKV, 128, 128, 0, 0);
  ln128_kernel<<<NKV, 128, 0, stream>>>(valb, a2[4], a2[5], lnkv);
  gemm(lnkv, wv2, nullptr, vhb, nullptr, nullptr, NKV, 128, 128, 0, 0);

  attn_core_kernel<<<64 * 4 * 2, 256, 0, stream>>>(
      qh2b, khb, vhb, nullptr, a2b, /*swapped=*/1, /*nQB=*/2, 1.0f);

  gemm(a2b, wo2, z2b, nullptr, a2[10], xskip, NS, 128, 128, /*remap*/2, 0);

  // MLP2 with residual
  ln128_kernel<<<NS, 128, 0, stream>>>(z2b, pre2_g, pre2_b, lntmp);
  gemm(lntmp, w12, nullptr, hbuf, m2b1, nullptr, NS, 256, 128, 0, /*gelu*/1);
  gemm(hbuf, w22, q2f, nullptr, m2b2, z2b, NS, 128, 256, 0, 0);

  // ------------- final LayerNorm, transposed store -------------
  final_ln_kernel<<<NS, 128, 0, stream>>>(q2f, post_g, post_b, (float*)d_out);
}
